// NDE_90220083020076
// MI455X (gfx1250) — matrix operand order repaired and checked by reading, not compiled, not run on hardware
//
#include <hip/hip_runtime.h>
#include <math.h>

typedef __attribute__((ext_vector_type(16))) _Float16 v16h;
typedef __attribute__((ext_vector_type(8)))  _Float16 v8h;
typedef __attribute__((ext_vector_type(8)))  float    v8f;

#define NB 65536
#define NSPR 8
#define NS (NB * NSPR)

__device__ __forceinline__ float clamp01f(float v) { return fminf(fmaxf(v, 0.f), 1.f); }

// D = A(16x32 f16) * B(32x16 f16) + C(16x16 f32)
__device__ __forceinline__ v8f wmma16(v16h a, v16h b, v8f c) {
  return __builtin_amdgcn_wmma_f32_16x16x32_f16(false, a, false, b, (short)0, c,
                                                false, false);
}

// A fragment: lane l holds row m=l&15; halves 0-7 = K kb..kb+7, halves 8-15 =
// K 16+kb..16+kb+7, kb=(l<16)?0:8 (ISA 7.12.2).  Both 8-half runs are
// contiguous & 16B-aligned in a row-major LDS tile -> 2x ds_load_b128 each.
__device__ __forceinline__ v16h load_a16(const _Float16* As, int lda, int k0) {
  const int l  = threadIdx.x & 31;
  const int m  = l & 15;
  const int kb = (l & 16) ? 8 : 0;
  const _Float16* p = As + m * lda + k0 + kb;
  v16h a;
  v8h lo = *(const v8h*)(p);
  v8h hi = *(const v8h*)(p + 16);
#pragma unroll
  for (int j = 0; j < 8; ++j) a[j] = lo[j];
#pragma unroll
  for (int j = 0; j < 8; ++j) a[8 + j] = hi[j];
  return a;
}

// B fragments are pre-swizzled in LDS: fragment (kt,nt) is 32 lanes x 16
// halves contiguous -> one 32B per-lane vector load (2x ds_load_b128).
__device__ __forceinline__ v16h load_bfrag(const _Float16* Wf, int ntiles, int kt,
                                           int nt) {
  const int l = threadIdx.x & 31;
  return *(const v16h*)(Wf + ((kt * ntiles + nt) * 32 + l) * 16);
}

// C/D fragment: lane l, VGPR i -> row i + ((l>=16)?8:0), col l&15
__device__ __forceinline__ void store_c_relu(_Float16* dst, int ldd, int n0, v8f c,
                                             const float* bias) {
  const int l   = threadIdx.x & 31;
  const int col = (l & 15) + n0;
  const int rb  = (l & 16) ? 8 : 0;
#pragma unroll
  for (int i = 0; i < 8; ++i) {
    float v = c[i] + bias[col];
    v = v > 0.f ? v : 0.f;
    dst[(rb + i) * ldd + col] = (_Float16)v;
  }
}

// Cooperative f32 -> f16 weight conversion directly into B-fragment order.
// Element for fragment (kt,nt), lane l, half j is W[kt*32+((l&16)?8:0)+(j&7)+
// ((j&8)?16:0)][nt*16+(l&15)], zero-padded outside rows x cols.
__device__ __forceinline__ void coop_w_frag(_Float16* dst, const float* src,
                                            int rows, int cols, int ktiles,
                                            int ntiles, int nthr, int tid) {
  const int total = ktiles * ntiles * 512;
  for (int i = tid; i < total; i += nthr) {
    const int f  = i >> 9;
    const int r  = i & 511;
    const int l  = r >> 4;
    const int j  = r & 15;
    const int kt = f / ntiles, nt = f % ntiles;
    const int k  = kt * 32 + ((l & 16) ? 8 : 0) + (j & 7) + ((j & 8) ? 16 : 0);
    const int n  = nt * 16 + (l & 15);
    float v = (k < rows && n < cols) ? src[k * cols + n] : 0.f;
    dst[i] = (_Float16)v;
  }
}

__device__ __forceinline__ void coop_b(float* dst, const float* src, int n, int nthr,
                                       int tid) {
  for (int i = tid; i < n; i += nthr) dst[i] = src[i];
}

// mip-triplane bilinear encode of one sample into a 64-wide f16 LDS row
// (48 real features + 16 zero pad), written as 16B packed stores.
__device__ __forceinline__ void encode_triplane_row(const float* __restrict__ grid,
                                                    float ux, float uy, float uz,
                                                    float lvlf, _Float16* dstRow) {
  int l0 = (int)lvlf;
  if (l0 > 3) l0 = 3;
  int l1 = (l0 + 1 > 3) ? 3 : l0 + 1;
  float wl = lvlf - (float)l0;
  float uu[3], vv[3];
  uu[0] = ux; vv[0] = uy;
  uu[1] = ux; vv[1] = uz;
  uu[2] = uy; vv[2] = uz;
#pragma unroll
  for (int pl = 0; pl < 3; ++pl) {
    float uf = clamp01f(uu[pl]) * 255.f;
    float vf = clamp01f(vv[pl]) * 255.f;
    int x0 = (int)uf, y0 = (int)vf;
    int x1 = (x0 + 1 > 255) ? 255 : x0 + 1;
    int y1 = (y0 + 1 > 255) ? 255 : y0 + 1;
    float wx = uf - (float)x0, wy = vf - (float)y0;
    float acc[16];
#pragma unroll
    for (int c = 0; c < 16; ++c) acc[c] = 0.f;
#pragma unroll
    for (int li = 0; li < 2; ++li) {
      int lvl = li ? l1 : l0;
      float lw = li ? wl : (1.f - wl);
      const float* g = grid + (size_t)(lvl * 3 + pl) * (256 * 256 * 16);
      const float4* g00 = (const float4*)(g + ((size_t)y0 * 256 + x0) * 16);
      const float4* g01 = (const float4*)(g + ((size_t)y0 * 256 + x1) * 16);
      const float4* g10 = (const float4*)(g + ((size_t)y1 * 256 + x0) * 16);
      const float4* g11 = (const float4*)(g + ((size_t)y1 * 256 + x1) * 16);
      float w00 = (1.f - wx) * (1.f - wy) * lw;
      float w01 = wx * (1.f - wy) * lw;
      float w10 = (1.f - wx) * wy * lw;
      float w11 = wx * wy * lw;
#pragma unroll
      for (int q = 0; q < 4; ++q) {
        float4 A = g00[q], Bv = g01[q], C = g10[q], D = g11[q];
        acc[q * 4 + 0] += A.x * w00 + Bv.x * w01 + C.x * w10 + D.x * w11;
        acc[q * 4 + 1] += A.y * w00 + Bv.y * w01 + C.y * w10 + D.y * w11;
        acc[q * 4 + 2] += A.z * w00 + Bv.z * w01 + C.z * w10 + D.z * w11;
        acc[q * 4 + 3] += A.w * w00 + Bv.w * w01 + C.w * w10 + D.w * w11;
      }
    }
    v8h p0, p1;
#pragma unroll
    for (int j = 0; j < 8; ++j) { p0[j] = (_Float16)acc[j]; p1[j] = (_Float16)acc[8 + j]; }
    *(v8h*)(dstRow + pl * 16)     = p0;
    *(v8h*)(dstRow + pl * 16 + 8) = p1;
  }
  v8h z = {};
  *(v8h*)(dstRow + 48) = z;
  *(v8h*)(dstRow + 56) = z;
}

// ---------------------------------------------------------------------------
// Kernel 1: per-sample triplane encode + sigma/radiance MLPs (WMMA) +
// wave-local transmittance + direct h_n / alpha_n stores.  256 samples/block.
// ---------------------------------------------------------------------------
__global__ __launch_bounds__(256) void nde_sample_kernel(
    const float* __restrict__ x, const float* __restrict__ wi,
    const float* __restrict__ rough, const float* __restrict__ t_starts,
    const float* __restrict__ t_ends, const int* __restrict__ ridx,
    const float* __restrict__ tri_grid, const float* __restrict__ tri_sigma,
    const float* __restrict__ nW1, const float* __restrict__ nB1,
    const float* __restrict__ nW2, const float* __restrict__ nB2,
    const float* __restrict__ nW3, const float* __restrict__ nB3,
    const float* __restrict__ gW1, const float* __restrict__ gB1,
    const float* __restrict__ gW2, const float* __restrict__ gB2,
    const float* __restrict__ gW3, const float* __restrict__ gB3,
    float* __restrict__ h_n, float* __restrict__ alpha_n) {
  __shared__ _Float16 sF[256 * 64];   // feature / hidden tile (32 KB)
  __shared__ _Float16 sW1[2 * 4 * 512];  // layer-1 fragments (8 KB)
  __shared__ _Float16 sW2[2 * 4 * 512];  // layer-2 fragments (8 KB)
  __shared__ _Float16 sW3[2 * 2 * 512];  // layer-3 fragments (4 KB)
  __shared__ float sB1[64], sB2[64], sB3[32];
  __shared__ float sVDT[256], sSDT[256], sWgt[256];

  const int tid      = threadIdx.x;
  const int lane     = tid & 31;
  const int waveBase = (tid >> 5) * 32;
  const int s        = blockIdx.x * 256 + tid;

  // ---- per-sample geometry ----
  const int ray  = ridx[s];
  const float ts = t_starts[s], te = t_ends[s];
  const float tm = 0.5f * (ts + te);
  const float dt = te - ts;
  const float rg = rough[ray];
  const float px = x[ray * 3 + 0] + tm * wi[ray * 3 + 0];
  const float py = x[ray * 3 + 1] + tm * wi[ray * 3 + 1];
  const float pz = x[ray * 3 + 2] + tm * wi[ray * 3 + 2];
  const float ux = (px + 1.f) * 0.5f;
  const float uy = (py + 1.f) * 0.5f;
  const float uz = (pz + 1.f) * 0.5f;
  const float rn = rg * rg * 1.73205080757f * tm * 0.5f;  // sqrt(T/(1-T)), T=0.75
  const bool valid = (ux >= 0.f && ux <= 1.f && uy >= 0.f && uy <= 1.f &&
                      uz >= 0.f && uz <= 1.f);
  sVDT[tid] = valid ? dt : 0.f;
  const float lvlf = fminf(fmaxf(log2f(1.f + rn * 256.f), 0.f), 3.f);

  __builtin_prefetch(tri_sigma, 0, 3);

  // ---- phase 0: sigma-branch weights (fragment-swizzled) + sigma features ----
  coop_w_frag(sW1, gW1, 48, 64, 2, 4, 256, tid);
  coop_w_frag(sW2, gW2, 64, 64, 2, 4, 256, tid);
  coop_w_frag(sW3, gW3, 64, 1, 2, 1, 256, tid);
  coop_b(sB1, gB1, 64, 256, tid);
  coop_b(sB2, gB2, 64, 256, tid);
  encode_triplane_row(tri_sigma, ux, uy, uz, lvlf, sF + tid * 64);
  __syncthreads();

  const float b3s = gB3[0];

  // ---- phase 1: sigma MLP (48->64->64->1) per wave over two 16-row tiles ----
  for (int t = 0; t < 2; ++t) {
    _Float16* Ar = sF + (waveBase + t * 16) * 64;
    v8f c0 = {}, c1 = {}, c2 = {}, c3 = {};
#pragma unroll
    for (int kt = 0; kt < 2; ++kt) {
      v16h a = load_a16(Ar, 64, kt * 32);
      c0 = wmma16(a, load_bfrag(sW1, 4, kt, 0), c0);
      c1 = wmma16(a, load_bfrag(sW1, 4, kt, 1), c1);
      c2 = wmma16(a, load_bfrag(sW1, 4, kt, 2), c2);
      c3 = wmma16(a, load_bfrag(sW1, 4, kt, 3), c3);
    }
    store_c_relu(Ar, 64, 0, c0, sB1);
    store_c_relu(Ar, 64, 16, c1, sB1);
    store_c_relu(Ar, 64, 32, c2, sB1);
    store_c_relu(Ar, 64, 48, c3, sB1);
    v8f d0 = {}, d1 = {}, d2 = {}, d3 = {};
#pragma unroll
    for (int kt = 0; kt < 2; ++kt) {
      v16h a = load_a16(Ar, 64, kt * 32);
      d0 = wmma16(a, load_bfrag(sW2, 4, kt, 0), d0);
      d1 = wmma16(a, load_bfrag(sW2, 4, kt, 1), d1);
      d2 = wmma16(a, load_bfrag(sW2, 4, kt, 2), d2);
      d3 = wmma16(a, load_bfrag(sW2, 4, kt, 3), d3);
    }
    store_c_relu(Ar, 64, 0, d0, sB2);
    store_c_relu(Ar, 64, 16, d1, sB2);
    store_c_relu(Ar, 64, 32, d2, sB2);
    store_c_relu(Ar, 64, 48, d3, sB2);
    v8f cs = {};
#pragma unroll
    for (int kt = 0; kt < 2; ++kt)
      cs = wmma16(load_a16(Ar, 64, kt * 32), load_bfrag(sW3, 1, kt, 0), cs);
    if ((lane & 15) == 0) {  // col 0 of the sigma output
      const int rb = waveBase + t * 16 + ((lane & 16) ? 8 : 0);
#pragma unroll
      for (int i = 0; i < 8; ++i) {
        float z  = cs[i] + b3s;
        float sp = (z > 20.f) ? z : log1pf(expf(z));  // softplus
        sSDT[rb + i] = sp * sVDT[rb + i];             // sigma*valid*dt
      }
    }
  }

  // ---- wave-local exclusive transmittance over each 8-sample ray ----
  {
    const int base = tid & ~7;
    const int j    = tid & 7;
    float excl = 0.f;
#pragma unroll
    for (int k = 0; k < 7; ++k)
      if (k < j) excl += sSDT[base + k];
    const float sdt = sSDT[tid];
    sWgt[tid] = expf(-excl) * (1.f - expf(-sdt));
    if (j == 0) {
      float tot = 0.f;
#pragma unroll
      for (int k = 0; k < 8; ++k) tot += sSDT[base + k];
      alpha_n[(blockIdx.x * 256 + base) >> 3] = 1.f - expf(-tot);  // sum of w
    }
  }
  __syncthreads();

  // ---- phase 2: radiance-branch weights + radiance features ----
  coop_w_frag(sW1, nW1, 48, 64, 2, 4, 256, tid);
  coop_w_frag(sW2, nW2, 64, 64, 2, 4, 256, tid);
  coop_w_frag(sW3, nW3, 64, 32, 2, 2, 256, tid);
  coop_b(sB1, nB1, 64, 256, tid);
  coop_b(sB2, nB2, 64, 256, tid);
  coop_b(sB3, nB3, 32, 256, tid);
  encode_triplane_row(tri_grid, ux, uy, uz, lvlf, sF + tid * 64);
  __syncthreads();

  // ---- phase 3: radiance MLP (48->64->64->32) + weighted ray accumulation ----
  for (int t = 0; t < 2; ++t) {
    _Float16* Ar = sF + (waveBase + t * 16) * 64;
    v8f c0 = {}, c1 = {}, c2 = {}, c3 = {};
#pragma unroll
    for (int kt = 0; kt < 2; ++kt) {
      v16h a = load_a16(Ar, 64, kt * 32);
      c0 = wmma16(a, load_bfrag(sW1, 4, kt, 0), c0);
      c1 = wmma16(a, load_bfrag(sW1, 4, kt, 1), c1);
      c2 = wmma16(a, load_bfrag(sW1, 4, kt, 2), c2);
      c3 = wmma16(a, load_bfrag(sW1, 4, kt, 3), c3);
    }
    store_c_relu(Ar, 64, 0, c0, sB1);
    store_c_relu(Ar, 64, 16, c1, sB1);
    store_c_relu(Ar, 64, 32, c2, sB1);
    store_c_relu(Ar, 64, 48, c3, sB1);
    v8f d0 = {}, d1 = {}, d2 = {}, d3 = {};
#pragma unroll
    for (int kt = 0; kt < 2; ++kt) {
      v16h a = load_a16(Ar, 64, kt * 32);
      d0 = wmma16(a, load_bfrag(sW2, 4, kt, 0), d0);
      d1 = wmma16(a, load_bfrag(sW2, 4, kt, 1), d1);
      d2 = wmma16(a, load_bfrag(sW2, 4, kt, 2), d2);
      d3 = wmma16(a, load_bfrag(sW2, 4, kt, 3), d3);
    }
    store_c_relu(Ar, 64, 0, d0, sB2);
    store_c_relu(Ar, 64, 16, d1, sB2);
    store_c_relu(Ar, 64, 32, d2, sB2);
    store_c_relu(Ar, 64, 48, d3, sB2);
#pragma unroll
    for (int nt = 0; nt < 2; ++nt) {
      v8f cc = {};
#pragma unroll
      for (int kt = 0; kt < 2; ++kt)
        cc = wmma16(load_a16(Ar, 64, kt * 32), load_bfrag(sW3, 2, kt, nt), cc);
      // rows i..i+7 of this C fragment form exactly one ray: reduce with weights
      const int col = (lane & 15) + nt * 16;
      const int rb  = waveBase + t * 16 + ((lane & 16) ? 8 : 0);
      float acc = 0.f;
#pragma unroll
      for (int i = 0; i < 8; ++i) acc += sWgt[rb + i] * (cc[i] + sB3[col]);
      h_n[((blockIdx.x * 256 + rb) >> 3) * 32 + col] = acc;  // no atomics needed
    }
  }
}

// ---------------------------------------------------------------------------
// Kernel 2: per-ray cubemap encode + blend + decode MLP (65->128->128->3).
// 64 rays / block (2 waves); W1/W2/W3 staged sequentially in one LDS buffer.
// ---------------------------------------------------------------------------
__global__ __launch_bounds__(64) void nde_ray_kernel(
    const float* __restrict__ wi, const float* __restrict__ rough,
    const float* __restrict__ fx, const float* __restrict__ wo,
    const float* __restrict__ cube_grid, const float* __restrict__ dW1,
    const float* __restrict__ dB1, const float* __restrict__ dW2,
    const float* __restrict__ dB2, const float* __restrict__ dW3,
    const float* __restrict__ dB3, const float* __restrict__ h_n,
    const float* __restrict__ alpha_n, float* __restrict__ out) {
  __shared__ _Float16 sA[64 * 128];      // activation tile (16 KB)
  __shared__ _Float16 sW[4 * 8 * 512];   // weight fragments (32 KB, reused)
  __shared__ float sB[128];

  const int tid      = threadIdx.x;
  const int lane     = tid & 31;
  const int waveBase = (tid >> 5) * 32;
  const int q        = blockIdx.x * 64 + tid;

  __builtin_prefetch(cube_grid, 0, 3);

  // ---- cubemap far-field feature ----
  const float d0 = wi[q * 3 + 0], d1 = wi[q * 3 + 1], d2 = wi[q * 3 + 2];
  const float a0 = fabsf(d0), a1 = fabsf(d1), a2 = fabsf(d2);
  int axis = 0;
  float m = a0;
  if (a1 > m) { axis = 1; m = a1; }
  if (a2 > m) { axis = 2; m = a2; }
  m += 1e-9f;
  const float dax = (axis == 0) ? d0 : ((axis == 1) ? d1 : d2);
  const float du  = (axis == 0) ? d1 : ((axis == 1) ? d2 : d0);
  const float dv  = (axis == 0) ? d2 : ((axis == 1) ? d0 : d1);
  const int face  = axis * 2 + (dax < 0.f ? 1 : 0);
  const float u   = (du / m) * 0.5f + 0.5f;
  const float v   = (dv / m) * 0.5f + 0.5f;
  const float lvlf = fminf(fmaxf(rough[q] * 3.f, 0.f), 3.f);
  int l0 = (int)lvlf;
  if (l0 > 3) l0 = 3;
  const int l1   = (l0 + 1 > 3) ? 3 : l0 + 1;
  const float wl = lvlf - (float)l0;
  const float uf = clamp01f(u) * 127.f, vf = clamp01f(v) * 127.f;
  const int x0 = (int)uf, y0 = (int)vf;
  const int x1 = (x0 + 1 > 127) ? 127 : x0 + 1;
  const int y1 = (y0 + 1 > 127) ? 127 : y0 + 1;
  const float wx = uf - (float)x0, wy = vf - (float)y0;
  float hf[32];
#pragma unroll
  for (int c = 0; c < 32; ++c) hf[c] = 0.f;
#pragma unroll
  for (int li = 0; li < 2; ++li) {
    const int lvl  = li ? l1 : l0;
    const float lw = li ? wl : (1.f - wl);
    const float* g = cube_grid + (size_t)(lvl * 6 + face) * (128 * 128 * 32);
    const float4* g00 = (const float4*)(g + ((size_t)y0 * 128 + x0) * 32);
    const float4* g01 = (const float4*)(g + ((size_t)y0 * 128 + x1) * 32);
    const float4* g10 = (const float4*)(g + ((size_t)y1 * 128 + x0) * 32);
    const float4* g11 = (const float4*)(g + ((size_t)y1 * 128 + x1) * 32);
    const float w00 = (1.f - wx) * (1.f - wy) * lw;
    const float w01 = wx * (1.f - wy) * lw;
    const float w10 = (1.f - wx) * wy * lw;
    const float w11 = wx * wy * lw;
#pragma unroll
    for (int qq = 0; qq < 8; ++qq) {
      float4 A = g00[qq], Bv = g01[qq], C = g10[qq], D = g11[qq];
      hf[qq * 4 + 0] += A.x * w00 + Bv.x * w01 + C.x * w10 + D.x * w11;
      hf[qq * 4 + 1] += A.y * w00 + Bv.y * w01 + C.y * w10 + D.y * w11;
      hf[qq * 4 + 2] += A.z * w00 + Bv.z * w01 + C.z * w10 + D.z * w11;
      hf[qq * 4 + 3] += A.w * w00 + Bv.w * w01 + C.w * w10 + D.w * w11;
    }
  }

  // ---- assemble decode-MLP input row: [fx | h_f*(1-a)+h_n | wo | pad] ----
  const float an = alpha_n[q];
  _Float16* row  = sA + tid * 128;
#pragma unroll
  for (int b8 = 0; b8 < 4; ++b8) {
    v8h pk;
#pragma unroll
    for (int j = 0; j < 8; ++j) pk[j] = (_Float16)fx[q * 32 + b8 * 8 + j];
    *(v8h*)(row + b8 * 8) = pk;
  }
#pragma unroll
  for (int b8 = 0; b8 < 4; ++b8) {
    v8h pk;
#pragma unroll
    for (int j = 0; j < 8; ++j) {
      int c = b8 * 8 + j;
      pk[j] = (_Float16)(hf[c] * (1.f - an) + h_n[q * 32 + c]);
    }
    *(v8h*)(row + 32 + b8 * 8) = pk;
  }
  {
    v8h pk = {};
    pk[0] = (_Float16)wo[q];
    *(v8h*)(row + 64) = pk;
    v8h z = {};
    *(v8h*)(row + 72) = z;
    *(v8h*)(row + 80) = z;
    *(v8h*)(row + 88) = z;
    *(v8h*)(row + 96) = z;
    *(v8h*)(row + 104) = z;
    *(v8h*)(row + 112) = z;
    *(v8h*)(row + 120) = z;
  }

  // ---- layer 1 (K=96 padded, N=128) ----
  coop_w_frag(sW, dW1, 65, 128, 3, 8, 64, tid);
  coop_b(sB, dB1, 128, 64, tid);
  __syncthreads();
  for (int t = 0; t < 2; ++t) {
    _Float16* Ar = sA + (waveBase + t * 16) * 128;
    v8f c[8];
#pragma unroll
    for (int nt = 0; nt < 8; ++nt) { v8f z = {}; c[nt] = z; }
#pragma unroll
    for (int kt = 0; kt < 3; ++kt) {
      v16h a = load_a16(Ar, 128, kt * 32);
#pragma unroll
      for (int nt = 0; nt < 8; ++nt)
        c[nt] = wmma16(a, load_bfrag(sW, 8, kt, nt), c[nt]);
    }
#pragma unroll
    for (int nt = 0; nt < 8; ++nt) store_c_relu(Ar, 128, nt * 16, c[nt], sB);
  }
  __syncthreads();

  // ---- layer 2 (K=128, N=128) ----
  coop_w_frag(sW, dW2, 128, 128, 4, 8, 64, tid);
  coop_b(sB, dB2, 128, 64, tid);
  __syncthreads();
  for (int t = 0; t < 2; ++t) {
    _Float16* Ar = sA + (waveBase + t * 16) * 128;
    v8f c[8];
#pragma unroll
    for (int nt = 0; nt < 8; ++nt) { v8f z = {}; c[nt] = z; }
#pragma unroll
    for (int kt = 0; kt < 4; ++kt) {
      v16h a = load_a16(Ar, 128, kt * 32);
#pragma unroll
      for (int nt = 0; nt < 8; ++nt)
        c[nt] = wmma16(a, load_bfrag(sW, 8, kt, nt), c[nt]);
    }
#pragma unroll
    for (int nt = 0; nt < 8; ++nt) store_c_relu(Ar, 128, nt * 16, c[nt], sB);
  }
  __syncthreads();

  // ---- layer 3 (K=128, N=3 padded to 16) + sigmoid + store ----
  coop_w_frag(sW, dW3, 128, 3, 4, 1, 64, tid);
  __syncthreads();
  for (int t = 0; t < 2; ++t) {
    _Float16* Ar = sA + (waveBase + t * 16) * 128;
    v8f cc = {};
#pragma unroll
    for (int kt = 0; kt < 4; ++kt)
      cc = wmma16(load_a16(Ar, 128, kt * 32), load_bfrag(sW, 1, kt, 0), cc);
    const int col = lane & 15;
    if (col < 3) {
      const int rb   = waveBase + t * 16 + ((lane & 16) ? 8 : 0);
      const float bb = dB3[col];
#pragma unroll
      for (int i = 0; i < 8; ++i) {
        float z = cc[i] + bb;
        out[(blockIdx.x * 64 + rb + i) * 3 + col] = 1.f / (1.f + expf(-z));
      }
    }
  }
}

extern "C" void kernel_launch(void* const* d_in, const int* in_sizes, int n_in,
                              void* d_out, int out_size, void* d_ws, size_t ws_size,
                              hipStream_t stream) {
  (void)in_sizes; (void)n_in; (void)out_size; (void)ws_size;
  const float* x     = (const float*)d_in[0];
  const float* wi    = (const float*)d_in[1];
  const float* rough = (const float*)d_in[2];
  const float* fx    = (const float*)d_in[3];
  const float* wo    = (const float*)d_in[4];
  const float* ts    = (const float*)d_in[5];
  const float* te    = (const float*)d_in[6];
  const int*   ri    = (const int*)d_in[7];
  const float* cube  = (const float*)d_in[8];
  const float* tri   = (const float*)d_in[9];
  const float* tris  = (const float*)d_in[10];
  const float* nW1 = (const float*)d_in[11];
  const float* nB1 = (const float*)d_in[12];
  const float* nW2 = (const float*)d_in[13];
  const float* nB2 = (const float*)d_in[14];
  const float* nW3 = (const float*)d_in[15];
  const float* nB3 = (const float*)d_in[16];
  const float* gW1 = (const float*)d_in[17];
  const float* gB1 = (const float*)d_in[18];
  const float* gW2 = (const float*)d_in[19];
  const float* gB2 = (const float*)d_in[20];
  const float* gW3 = (const float*)d_in[21];
  const float* gB3 = (const float*)d_in[22];
  const float* dW1 = (const float*)d_in[23];
  const float* dB1 = (const float*)d_in[24];
  const float* dW2 = (const float*)d_in[25];
  const float* dB2 = (const float*)d_in[26];
  const float* dW3 = (const float*)d_in[27];
  const float* dB3 = (const float*)d_in[28];

  float* ws_hn = (float*)d_ws;                 // [B,32] near-field accumulation
  float* ws_an = ws_hn + (size_t)NB * 32;      // [B]    opacity

  nde_sample_kernel<<<NS / 256, 256, 0, stream>>>(
      x, wi, rough, ts, te, ri, tri, tris,
      nW1, nB1, nW2, nB2, nW3, nB3,
      gW1, gB1, gW2, gB2, gW3, gB3,
      ws_hn, ws_an);

  nde_ray_kernel<<<NB / 64, 64, 0, stream>>>(
      wi, rough, fx, wo, cube,
      dW1, dB1, dW2, dB2, dW3, dB3,
      ws_hn, ws_an, (float*)d_out);
}
